// MyGraphModel_83820581748942
// MI455X (gfx1250) — matrix operand order repaired and checked by reading, not compiled, not run on hardware
//
#include <hip/hip_runtime.h>
#include <hip/hip_bf16.h>

// ---------------------------------------------------------------------------
// GraphSAGE 2-layer forward for MI455X (gfx1250, wave32, WMMA).
//   layer: h = relu( mean_agg(x) @ W1l.T + b1l + x @ W1r.T )
//   out   =        mean_agg(h) @ W2l.T + b2l + h @ W2r.T
// Aggregation: f32 global atomics (sums are L2-resident: 51/102 MB < 192 MB L2).
// GEMMs: v_wmma_f32_16x16x32_bf16, f32 accumulate.
//   - Weights pre-converted to bf16 once (256 KB) -> B fragments load raw,
//     zero conversion VALU in the hot loop.
//   - A fragments converted f32->bf16 via compiler cast (HW pk-cvt if present).
//   - mean-divide / bias / relu fused into the GEMM.
// ---------------------------------------------------------------------------

typedef __attribute__((ext_vector_type(16))) __bf16 v16bf;
typedef __attribute__((ext_vector_type(8)))  float  v8f;

// A-matrix fragment (16x32 bf16), contiguous-K layout (same order as B):
//  lane L: m = L&15, half = L>>4
//  vector elts 0..15 = K(k0 + half*16 + 0..15)
__device__ __forceinline__ v16bf load_fragA(const float* __restrict__ base, int ld,
                                            int k0, int lane, float scale) {
    const int m    = lane & 15;
    const int half = lane >> 4;
    const float* p = base + (long long)m * ld + k0 + half * 16;
    float v[16];
    *(float4*)(v + 0)  = *(const float4*)(p + 0);
    *(float4*)(v + 4)  = *(const float4*)(p + 4);
    *(float4*)(v + 8)  = *(const float4*)(p + 8);
    *(float4*)(v + 12) = *(const float4*)(p + 12);
    v16bf f;
#pragma unroll
    for (int i = 0; i < 16; ++i) f[i] = (__bf16)(v[i] * scale);
    return f;
}

// B-matrix fragment (32x16 bf16), K-grouped layout, loaded raw from the
// pre-converted bf16 weight copy:
//  lane L: n = L&15, kg = L>>4 ; vector elt i = K(k0 + kg*16 + i)
//  B[k][n] = W[n][k]  (W row-major [outC][K], 32B-aligned rows)
__device__ __forceinline__ v16bf load_fragB(const __bf16* __restrict__ w, int ld,
                                            int k0, int lane) {
    const int n  = lane & 15;
    const int kg = lane >> 4;
    return *(const v16bf*)(w + (long long)n * ld + k0 + kg * 16);
}

// ---------------------------------------------------------------------------
__global__ void zero_kernel(float* __restrict__ p, long long nFloats) {
    long long i = ((long long)blockIdx.x * blockDim.x + threadIdx.x) * 4;
    if (i < nFloats) *(float4*)(p + i) = float4{0.f, 0.f, 0.f, 0.f};
}

__global__ void w2bf_kernel(const float* __restrict__ w, __bf16* __restrict__ o, int n) {
    int i = blockIdx.x * blockDim.x + threadIdx.x;
    if (i < n) o[i] = (__bf16)w[i];
}

__global__ void degree_kernel(const long long* __restrict__ ei,
                              float* __restrict__ cnt, long long E) {
    long long e = (long long)blockIdx.x * blockDim.x + threadIdx.x;
    if (e < E) atomicAdd(&cnt[ei[E + e]], 1.0f);
}

// One thread = one (edge, 4-channel group). 32/64 consecutive lanes cover one
// full source row -> coalesced gather; atomic adds land in L2-resident sums.
template <int C, int LOGC4>
__global__ void scatter_kernel(const float* __restrict__ feat,
                               const long long* __restrict__ ei,
                               long long E, float* __restrict__ sum) {
    long long tid = (long long)blockIdx.x * blockDim.x + threadIdx.x;
    if (tid >= (E << LOGC4)) return;
    long long e = tid >> LOGC4;
    int g = (int)(tid & ((1 << LOGC4) - 1));
    long long s = ei[e];
    long long d = ei[E + e];
    float4 v = *(const float4*)(feat + s * C + g * 4);
    float* o = sum + d * C + g * 4;
    atomicAdd(o + 0, v.x);
    atomicAdd(o + 1, v.y);
    atomicAdd(o + 2, v.z);
    atomicAdd(o + 3, v.w);
}

// Fused SAGE linear:  out = act( (Asum/deg) @ Wl.T + Aroot @ Wr.T + bias )
// Wave tile: 16 rows x 64 cols (4 x 16x16 accumulators, A reused 4x).
// Block = 64 threads = 2 waves (stacked M-tiles). Requires nrows % 32 == 0.
template <int K, int NCOLS, bool RELU>
__global__ void sage_gemm(const float* __restrict__ Asum,
                          const float* __restrict__ cnt,
                          const float* __restrict__ Aroot,
                          const __bf16* __restrict__ Wl,
                          const __bf16* __restrict__ Wr,
                          const float* __restrict__ bias,
                          float* __restrict__ out, int nrows) {
    const int lane = threadIdx.x & 31;
    const int wave = threadIdx.x >> 5;
    const int row0 = (blockIdx.y * 2 + wave) * 16;
    const int col0 = blockIdx.x * 64;
    if (row0 + 16 > nrows) return;          // wave-uniform

    const int m    = lane & 15;
    const int half = lane >> 4;
    const float sc = 1.0f / fmaxf(cnt[row0 + m], 1.0f);   // fused mean divide

    v8f acc[4];
#pragma unroll
    for (int t = 0; t < 4; ++t)
#pragma unroll
        for (int r = 0; r < 8; ++r) acc[t][r] = 0.0f;

    const float* a1base = Asum  + (long long)row0 * K;
    const float* a2base = Aroot + (long long)row0 * K;

    for (int k0 = 0; k0 < K; k0 += 32) {
        v16bf a1 = load_fragA(a1base, K, k0, lane, sc);
        v16bf a2 = load_fragA(a2base, K, k0, lane, 1.0f);
#pragma unroll
        for (int t = 0; t < 4; ++t) {
            v16bf b1 = load_fragB(Wl + (long long)(col0 + t * 16) * K, K, k0, lane);
            v16bf b2 = load_fragB(Wr + (long long)(col0 + t * 16) * K, K, k0, lane);
            acc[t] = __builtin_amdgcn_wmma_f32_16x16x32_bf16(
                false, a1, false, b1, (short)0, acc[t], false, false);
            acc[t] = __builtin_amdgcn_wmma_f32_16x16x32_bf16(
                false, a2, false, b2, (short)0, acc[t], false, false);
        }
    }

    // C/D layout: lane L -> n = L&15 ; VGPR r -> row = r + 8*(L>>4)
#pragma unroll
    for (int t = 0; t < 4; ++t) {
        const int n = col0 + t * 16 + m;
        const float b = bias[n];
#pragma unroll
        for (int r = 0; r < 8; ++r) {
            const int mr = row0 + r + 8 * half;
            float v = acc[t][r] + b;
            if (RELU) v = fmaxf(v, 0.0f);
            out[(long long)mr * NCOLS + n] = v;
        }
    }
}

// ---------------------------------------------------------------------------
extern "C" void kernel_launch(void* const* d_in, const int* in_sizes, int n_in,
                              void* d_out, int out_size, void* d_ws, size_t ws_size,
                              hipStream_t stream) {
    const float*     x   = (const float*)d_in[0];
    const long long* ei  = (const long long*)d_in[1];   // int64 edge_index [2,E]
    const float*     W1l = (const float*)d_in[2];       // [256,128]
    const float*     b1l = (const float*)d_in[3];       // [256]
    const float*     W1r = (const float*)d_in[4];       // [256,128]
    const float*     W2l = (const float*)d_in[5];       // [128,256]
    const float*     b2l = (const float*)d_in[6];       // [128]
    const float*     W2r = (const float*)d_in[7];       // [128,256]
    float*           out = (float*)d_out;

    const long long N = in_sizes[0] / 128;   // 100000
    const long long E = in_sizes[1] / 2;     // 800000
    const int WN = 256 * 128;                // elements per weight matrix

    // workspace layout:
    //   bf16 weights (4 x 32768 elts, 32B-aligned) | cnt[N] | sum1[N*128]
    //   | sum2[N*256] | h[N*256]                       (all f32)
    __bf16* w1l_bf = (__bf16*)d_ws;
    __bf16* w1r_bf = w1l_bf + WN;
    __bf16* w2l_bf = w1r_bf + WN;
    __bf16* w2r_bf = w2l_bf + WN;
    float*  cnt    = (float*)(w2r_bf + WN);
    float*  sum1   = cnt + N;
    float*  sum2   = sum1 + N * 128;
    float*  h      = sum2 + N * 256;

    // 1) zero cnt + sum1 + sum2 (deterministic re-init every call)
    {
        long long zf = N * (1 + 128 + 256);
        long long thr = (zf + 3) / 4;
        zero_kernel<<<dim3((unsigned)((thr + 255) / 256)), 256, 0, stream>>>(cnt, zf);
    }
    // 2) pre-convert weights to bf16 (tiny, L2-resident afterwards)
    {
        dim3 g((WN + 255) / 256);
        w2bf_kernel<<<g, 256, 0, stream>>>(W1l, w1l_bf, WN);
        w2bf_kernel<<<g, 256, 0, stream>>>(W1r, w1r_bf, WN);
        w2bf_kernel<<<g, 256, 0, stream>>>(W2l, w2l_bf, WN);
        w2bf_kernel<<<g, 256, 0, stream>>>(W2r, w2r_bf, WN);
    }
    // 3) in-degrees
    degree_kernel<<<dim3((unsigned)((E + 255) / 256)), 256, 0, stream>>>(ei, cnt, E);
    // 4) layer-1 scatter: sum1[dst] += x[src]
    scatter_kernel<128, 5><<<dim3((unsigned)((E * 32 + 255) / 256)), 256, 0, stream>>>(
        x, ei, E, sum1);
    // 5) h = relu( (sum1/deg)@W1l.T + b1l + x@W1r.T )   [N,256]
    sage_gemm<128, 256, true><<<dim3(4, (unsigned)(N / 32)), 64, 0, stream>>>(
        sum1, cnt, x, w1l_bf, w1r_bf, b1l, h, (int)N);
    // 6) layer-2 scatter: sum2[dst] += h[src]
    scatter_kernel<256, 6><<<dim3((unsigned)((E * 64 + 255) / 256)), 256, 0, stream>>>(
        h, ei, E, sum2);
    // 7) out = (sum2/deg)@W2l.T + b2l + h@W2r.T   [N,128]
    sage_gemm<256, 128, false><<<dim3(2, (unsigned)(N / 32)), 64, 0, stream>>>(
        sum2, cnt, h, w2l_bf, w2r_bf, b2l, out, (int)N);
}
